// DuetL_7052336300566
// MI455X (gfx1250) — matrix operand order repaired and checked by reading, not compiled, not run on hardware
//
#include <hip/hip_runtime.h>
#include <hip/hip_bf16.h>

// ---------------------------------------------------------------------------
// Types for WMMA
// ---------------------------------------------------------------------------
typedef __attribute__((ext_vector_type(16))) __bf16 v16bf;
typedef __attribute__((ext_vector_type(16))) short  v16s;
typedef __attribute__((ext_vector_type(8)))  float  v8f;
typedef __attribute__((ext_vector_type(4)))  float  v4f;
typedef __attribute__((ext_vector_type(4)))  unsigned int u32x4;

union Frag { u32x4 q[2]; v16s s; };   // 32 bytes = 16 bf16 fragment

#define DEV static __device__ __forceinline__

DEV unsigned short f2bf(float f) {
  unsigned int u = __builtin_bit_cast(unsigned int, f);
  u += 0x7fffu + ((u >> 16) & 1u);          // round-to-nearest-even
  return (unsigned short)(u >> 16);
}

// ---------------------------------------------------------------------------
// f32 -> bf16 conversion (flat)
// ---------------------------------------------------------------------------
__global__ void k_f32_to_bf16(const float* __restrict__ src,
                              unsigned short* __restrict__ dst, int n) {
  int i = blockIdx.x * blockDim.x + threadIdx.x;
  if (i < n) dst[i] = f2bf(src[i]);
}

// Stem input: NCHW f32 [128,3,32,32] -> NHWC bf16 [128,32,32,32] (C padded to 32)
__global__ void k_stem_nhwc(const float* __restrict__ src,
                            unsigned short* __restrict__ dst) {
  int i = blockIdx.x * blockDim.x + threadIdx.x;   // over 128*1024*32
  if (i >= 128 * 1024 * 32) return;
  int c  = i & 31;
  int sp = i >> 5;                                  // n*1024 + y*32 + x
  unsigned short v = 0;
  if (c < 3) v = f2bf(src[((sp >> 10) * 3 + c) * 1024 + (sp & 1023)]);
  dst[i] = v;
}

// Conv weights: w[co][ci][kk] f32 -> wT[co][kk][ci'] bf16, ci padded to Cinp.
__global__ void k_w_conv_bf16(const float* __restrict__ src,
                              unsigned short* __restrict__ dst,
                              int Cout, int Cin, int Cinp, int KHW2) {
  int i = blockIdx.x * blockDim.x + threadIdx.x;
  if (i >= Cout * KHW2 * Cinp) return;
  int ci = i % Cinp;
  int t  = i / Cinp;            // co*KHW2 + kk
  int kk = t % KHW2;
  int co = t / KHW2;
  dst[i] = (ci < Cin) ? f2bf(src[(co * Cin + ci) * KHW2 + kk]) : (unsigned short)0;
}

// f32 [E][R][Cc] -> bf16 [E][Cc][R]  (transpose inner 2 dims)
__global__ void k_transpose_bf16(const float* __restrict__ src,
                                 unsigned short* __restrict__ dst,
                                 int E, int R, int Cc) {
  int i = blockIdx.x * blockDim.x + threadIdx.x;
  int total = E * R * Cc;
  if (i >= total) return;
  int e = i / (R * Cc);
  int r2 = i - e * R * Cc;
  int r = r2 / Cc;
  int c = r2 - r * Cc;
  dst[(e * Cc + c) * R + r] = f2bf(src[i]);
}

// ---------------------------------------------------------------------------
// NHWC implicit-GEMM convolution as 9 (or 1) shifted GEMMs.
// One wave32 computes a 64x64 (Cout x spatial) tile: 16 accumulators,
// 8 A-loads + 8 B-loads per 16 WMMAs. All loads b128; one boundary predicate
// per lane per filter tap. Cout must be a multiple of 64 (true for ResNet).
//   inBf: [N,Hin,Win,Cinp] bf16 (channel-padded), wBf: [Cout,KHW2,Cinp] bf16,
//   out:  [N,Ho,Wo,Cout] f32.
// ---------------------------------------------------------------------------
__global__ __launch_bounds__(256)
void k_conv_wmma(const unsigned short* __restrict__ inBf,
                 const unsigned short* __restrict__ wBf,
                 float* __restrict__ out,
                 int N, int Cinp, int Hin, int Win,
                 int Cout, int Ho, int Wo,
                 int stride, int pad, int KHW) {
  const int HoWo = Ho * Wo;
  const int G = (N * HoWo) >> 6;           // groups of 4 N-tiles (64 cols)
  int lane = threadIdx.x & 31;
  int wave = threadIdx.x >> 5;
  int tg = blockIdx.x * 8 + wave;
  int tm = blockIdx.y;                     // 64-row (Cout) tile
  if (tg >= G) return;                     // wave-uniform exit
  int half = lane >> 4;
  int l15  = lane & 15;
  const int KHW2 = KHW * KHW;
  const int wstride = KHW2 * Cinp;

  int nimg[4], oya[4], oxa[4];
#pragma unroll
  for (int j = 0; j < 4; ++j) {
    int ncol = (tg << 6) + (j << 4) + l15;
    int ni = ncol / HoWo;
    int rem = ncol - ni * HoWo;
    int oy = rem / Wo;
    nimg[j] = ni; oya[j] = oy; oxa[j] = rem - oy * Wo;
  }

  v8f acc[16] = {};                        // acc[mi*4 + j]
  const unsigned short* arow[4];
#pragma unroll
  for (int mi = 0; mi < 4; ++mi)
    arow[mi] = wBf + (size_t)((tm << 6) + (mi << 4) + l15) * wstride + (half << 4);

  for (int kk = 0; kk < KHW2; ++kk) {
    int ky = kk / KHW;
    int kx = kk - ky * KHW;
    // prefetch next tap's weight slice (block-wide shared operand)
    if (kk + 1 < KHW2) {
#pragma unroll
      for (int mi = 0; mi < 4; ++mi)
        __builtin_prefetch(arow[mi] + (kk + 1) * Cinp, 0, 3);
    }
    int basej[4]; bool okj[4];
#pragma unroll
    for (int j = 0; j < 4; ++j) {
      int iy = oya[j] * stride - pad + ky;
      int ix = oxa[j] * stride - pad + kx;
      okj[j] = (iy >= 0) & (iy < Hin) & (ix >= 0) & (ix < Win);
      basej[j] = ((nimg[j] * Hin + iy) * Win + ix) * Cinp + (half << 4);
    }
    for (int k0 = 0; k0 < Cinp; k0 += 32) {
      Frag fa[4];
#pragma unroll
      for (int mi = 0; mi < 4; ++mi) {
        const u32x4* ap = (const u32x4*)(arow[mi] + kk * Cinp + k0);
        fa[mi].q[0] = ap[0];
        fa[mi].q[1] = ap[1];
      }
#pragma unroll
      for (int j = 0; j < 4; ++j) {
        Frag fb;
        if (okj[j]) {
          const u32x4* bp = (const u32x4*)(inBf + basej[j] + k0);
          fb.q[0] = bp[0];
          fb.q[1] = bp[1];
        } else {
          fb.q[0] = (u32x4){0, 0, 0, 0};
          fb.q[1] = (u32x4){0, 0, 0, 0};
        }
        v16bf bv = __builtin_bit_cast(v16bf, fb.s);
#pragma unroll
        for (int mi = 0; mi < 4; ++mi) {
          acc[mi * 4 + j] = __builtin_amdgcn_wmma_f32_16x16x32_bf16(
              false, __builtin_bit_cast(v16bf, fa[mi].s), false, bv,
              (short)0, acc[mi * 4 + j], false, false);
        }
      }
    }
  }

  // NHWC store: channels are the fast axis -> contiguous b128 stores.
#pragma unroll
  for (int j = 0; j < 4; ++j) {
    float* opb = out + (size_t)((nimg[j] * Ho + oya[j]) * Wo + oxa[j]) * Cout
                     + (tm << 6) + (half << 3);
#pragma unroll
    for (int mi = 0; mi < 4; ++mi) {
      v8f a = acc[mi * 4 + j];
      float* op = opb + (mi << 4);
      v4f lo = {a[0], a[1], a[2], a[3]};
      v4f hi = {a[4], a[5], a[6], a[7]};
      *(v4f*)op = lo;
      *((v4f*)op + 1) = hi;
    }
  }
}

// ---------------------------------------------------------------------------
// Dense bf16 GEMM, transposed-B only: C[M,N] = A[M,K] x B(n,k)[N,K]^T.
// One wave computes 64x64 (16 accumulators); all fragment loads b128.
// Requires: K % 32 == 0, rows 16B aligned, N % 64 == 0. M-edge guarded.
// ---------------------------------------------------------------------------
__global__ __launch_bounds__(256)
void k_gemm_wmma(const unsigned short* __restrict__ A, int lda,
                 const unsigned short* __restrict__ B, int ldb,
                 const float* __restrict__ bias, int relu,
                 float* __restrict__ C, int ldc,
                 int M, int N, int K) {
  int lane = threadIdx.x & 31;
  int wave = threadIdx.x >> 5;
  int tilesM = (M + 63) >> 6;
  int G = N >> 6;
  int widx = blockIdx.x * 8 + wave;
  if (widx >= tilesM * G) return;
  int tm = widx / G;
  int tg = widx - tm * G;
  int half = lane >> 4;
  int l15  = lane & 15;

  bool mok[4];
  const unsigned short* arow[4];
#pragma unroll
  for (int mi = 0; mi < 4; ++mi) {
    int Mrow = (tm << 6) + (mi << 4) + l15;
    mok[mi] = Mrow < M;
    arow[mi] = A + (size_t)Mrow * lda + (half << 4);
  }

  v8f acc[16] = {};
  for (int k0 = 0; k0 < K; k0 += 32) {
    Frag fa[4];
#pragma unroll
    for (int mi = 0; mi < 4; ++mi) {
      if (mok[mi]) {
        const u32x4* ap = (const u32x4*)(arow[mi] + k0);
        fa[mi].q[0] = ap[0];
        fa[mi].q[1] = ap[1];
      } else {
        fa[mi].q[0] = (u32x4){0, 0, 0, 0};
        fa[mi].q[1] = (u32x4){0, 0, 0, 0};
      }
    }
#pragma unroll
    for (int j = 0; j < 4; ++j) {
      int ncol = (tg << 6) + (j << 4) + l15;
      Frag fb;
      const u32x4* bp = (const u32x4*)(B + (size_t)ncol * ldb + k0 + (half << 4));
      fb.q[0] = bp[0];
      fb.q[1] = bp[1];
      v16bf bv = __builtin_bit_cast(v16bf, fb.s);
#pragma unroll
      for (int mi = 0; mi < 4; ++mi) {
        acc[mi * 4 + j] = __builtin_amdgcn_wmma_f32_16x16x32_bf16(
            false, __builtin_bit_cast(v16bf, fa[mi].s), false, bv,
            (short)0, acc[mi * 4 + j], false, false);
      }
    }
  }
#pragma unroll
  for (int j = 0; j < 4; ++j) {
    int ncol = (tg << 6) + (j << 4) + l15;
#pragma unroll
    for (int mi = 0; mi < 4; ++mi) {
      v8f a = acc[mi * 4 + j];
#pragma unroll
      for (int v = 0; v < 8; ++v) {
        int m = (tm << 6) + (mi << 4) + v + (half << 3);
        if (m < M) {
          float x = a[v];
          if (bias) x += bias[m];
          if (relu) x = fmaxf(x, 0.f);
          C[(size_t)m * ldc + ncol] = x;
        }
      }
    }
  }
}

// ---------------------------------------------------------------------------
// Training-mode BN batch stats (biased variance), NHWC, one block per channel.
// ---------------------------------------------------------------------------
__global__ void k_bn_stats(const float* __restrict__ x, int cnt, int C,
                           float* __restrict__ mean, float* __restrict__ rstd) {
  int c = blockIdx.x;
  float s = 0.f, s2 = 0.f;
  for (int t = threadIdx.x; t < cnt; t += 256) {
    float v = x[(size_t)t * C + c];
    s += v; s2 += v * v;
  }
  __shared__ float ss[256], ss2[256];
  ss[threadIdx.x] = s; ss2[threadIdx.x] = s2;
  __syncthreads();
  for (int o = 128; o > 0; o >>= 1) {
    if (threadIdx.x < o) {
      ss[threadIdx.x]  += ss[threadIdx.x + o];
      ss2[threadIdx.x] += ss2[threadIdx.x + o];
    }
    __syncthreads();
  }
  if (threadIdx.x == 0) {
    float m = ss[0] / cnt;
    float var = ss2[0] / cnt - m * m;
    mean[c] = m;
    rstd[c] = rsqrtf(var + 1e-5f);
  }
}

// BN apply NHWC (+optional residual, +relu), dual f32 / bf16 output. C = pow2.
__global__ void k_bn_apply(const float* __restrict__ x, int total, int Cmask,
                           const float* __restrict__ g, const float* __restrict__ b,
                           const float* __restrict__ mean, const float* __restrict__ rstd,
                           const float* __restrict__ resid, int relu,
                           float* __restrict__ outF, unsigned short* __restrict__ outBf) {
  int i = blockIdx.x * blockDim.x + threadIdx.x;
  if (i >= total) return;
  int c = i & Cmask;
  float v = g[c] * (x[i] - mean[c]) * rstd[c] + b[c];
  if (resid) v += resid[i];
  if (relu)  v = fmaxf(v, 0.f);
  if (outF)  outF[i]  = v;
  if (outBf) outBf[i] = f2bf(v);
}

// Global average pool NHWC, one thread per (n,c).
__global__ void k_avgpool(const float* __restrict__ x, int C, int HW,
                          float* __restrict__ z, int Nn) {
  int i = blockIdx.x * blockDim.x + threadIdx.x;
  if (i >= Nn * C) return;
  int n = i / C;
  int c = i - n * C;
  float s = 0.f;
  for (int t = 0; t < HW; ++t) s += x[(size_t)(n * HW + t) * C + c];
  z[i] = s / HW;
}

// Column softmax entropy: P is [C][Ntot] (GEMM output), H per column.
__global__ void k_entropy(const float* __restrict__ P, int C, int Ntot,
                          float* __restrict__ H) {
  int n = blockIdx.x * blockDim.x + threadIdx.x;
  if (n >= Ntot) return;
  float mx = -3.4e38f;
  for (int c = 0; c < C; ++c) mx = fmaxf(mx, P[c * Ntot + n]);
  float se = 0.f;
  for (int c = 0; c < C; ++c) se += __expf(P[c * Ntot + n] - mx);
  float h = 0.f;
  for (int c = 0; c < C; ++c) {
    float p = __expf(P[c * Ntot + n] - mx) / se;
    h -= p * __logf(fmaxf(p, 1e-9f));
  }
  H[n] = h;
}

// Lower median of 256 entropies (bitonic sort in LDS) -> n_exp in {3,4}.
__global__ void k_median_n(const float* __restrict__ H, int* __restrict__ nOut) {
  __shared__ float s[256];
  int t = threadIdx.x;
  s[t] = H[t];
  __syncthreads();
  for (int ksz = 2; ksz <= 256; ksz <<= 1) {
    for (int j = ksz >> 1; j > 0; j >>= 1) {
      int ixj = t ^ j;
      if (ixj > t) {
        bool up = ((t & ksz) == 0);
        float a = s[t], b = s[ixj];
        bool sw = up ? (a > b) : (a < b);
        if (sw) { s[t] = b; s[ixj] = a; }
      }
      __syncthreads();
    }
  }
  float tau = s[127];                 // (256-1)//2, ascending sort
  nOut[t] = (H[t] <= tau) ? 3 : 4;
}

// Greedy diverse-expert gate selection + gated logit mix, one thread/sample.
// L layout: [K][C][Bn] (stacked per-expert GEMM outputs).
__global__ void k_gates_mix(const float* __restrict__ L,
                            const int* __restrict__ nArr,
                            float* __restrict__ out,
                            int Bn, int Kex, int C) {
  int b = blockIdx.x * blockDim.x + threadIdx.x;
  if (b >= Bn) return;
  float conf[6], sumE[6], mxl[6], invnrm[6];
  for (int k = 0; k < Kex; ++k) {
    const float* Lk = L + (size_t)(k * C) * Bn + b;
    float mx = -3.4e38f;
    for (int c = 0; c < C; ++c) mx = fmaxf(mx, Lk[c * Bn]);
    float se = 0.f, se2 = 0.f;
    for (int c = 0; c < C; ++c) {
      float e = __expf(Lk[c * Bn] - mx);
      se += e; se2 += e * e;
    }
    mxl[k] = mx; sumE[k] = se;
    conf[k] = 1.f / se;                       // max softmax prob
    float nrm = sqrtf(se2) / se;              // ||p||_2
    invnrm[k] = 1.f / fmaxf(nrm, 1e-8f);
  }
  float S[6][6];
  for (int k = 0; k < Kex; ++k)
    for (int j = k; j < Kex; ++j) {
      const float* Lk = L + (size_t)(k * C) * Bn + b;
      const float* Lj = L + (size_t)(j * C) * Bn + b;
      float d = 0.f;
      for (int c = 0; c < C; ++c)
        d += __expf(Lk[c * Bn] - mxl[k]) * __expf(Lj[c * Bn] - mxl[j]);
      d /= (sumE[k] * sumE[j]);
      float v = d * invnrm[k] * invnrm[j];
      S[k][j] = v; S[j][k] = v;
    }
  int best0 = 0; float bc = conf[0];
  for (int k = 1; k < Kex; ++k) if (conf[k] > bc) { bc = conf[k]; best0 = k; }
  int sel = 1 << best0;
  int nsel = nArr[b];
  for (int i = 1; i <= 3; ++i) {
    float bd = -3.4e38f; int bk = 0;
    for (int k = 0; k < Kex; ++k) {
      float dist;
      if (sel & (1 << k)) dist = -3.4e38f;
      else {
        float ms = -3.4e38f;
        for (int j = 0; j < Kex; ++j)
          if (sel & (1 << j)) ms = fmaxf(ms, S[k][j]);
        dist = 1.f - ms;
      }
      if (dist > bd) { bd = dist; bk = k; }
    }
    if (i < nsel) sel |= 1 << bk;
  }
  float gl[6], gm = -3.4e38f;
  for (int k = 0; k < Kex; ++k) {
    float c0 = (sel & (1 << k)) ? conf[k] : -10000.f;
    gl[k] = c0 * 5.0f;
    gm = fmaxf(gm, gl[k]);
  }
  float gs = 0.f;
  for (int k = 0; k < Kex; ++k) { gl[k] = __expf(gl[k] - gm); gs += gl[k]; }
  for (int k = 0; k < Kex; ++k) gl[k] /= gs;
  for (int c = 0; c < C; ++c) {
    float o = 0.f;
    for (int k = 0; k < Kex; ++k) o += L[(size_t)(k * C + c) * Bn + b] * gl[k];
    out[b * C + c] = o;
  }
}

// ---------------------------------------------------------------------------
// Host side
// ---------------------------------------------------------------------------
struct Blk { const float *w1,*g1,*b1,*w2,*g2,*b2,*wd,*gd,*bd; int ds; };
struct Net {
  const float *x_u,*x_r,*conv1_w,*bn1_g,*bn1_b,*probe_w,*probe_b;
  const float *uW1,*ub1,*uW2,*ub2,*rW1,*rb1,*rW2,*rb2;
  Blk blk[8];
};

static const int BLK_CIN[8]  = {64,64,64,128,128,256,256,512};
static const int BLK_COUT[8] = {64,64,128,128,256,256,512,512};
static const int BLK_S[8]    = {1,1,2,1,2,1,2,1};

// Map flat input list to named params; auto-detect pytree flattening order.
static void map_inputs(void* const* d_in, const int* in_sizes, int n_in, Net& P) {
  (void)n_in;
  auto F = [&](int i) { return (const float*)d_in[i]; };
  bool xFirst = (in_sizes[0] == 128 * 3 * 32 * 32);
  int i = 0;
  if (xFirst) { P.x_u = F(0); P.x_r = F(1); i = 2; }
  bool insertion = xFirst && (in_sizes[i] == 1728);  // conv1_w first => dict order
  if (insertion) {
    P.conv1_w = F(i++); P.bn1_g = F(i++); P.bn1_b = F(i++);
    for (int b = 0; b < 8; ++b) {
      Blk& K = P.blk[b];
      K.ds = (BLK_S[b] != 1) || (BLK_CIN[b] != BLK_COUT[b]);
      K.w1 = F(i++); K.g1 = F(i++); K.b1 = F(i++);
      K.w2 = F(i++); K.g2 = F(i++); K.b2 = F(i++);
      if (K.ds) { K.wd = F(i++); K.gd = F(i++); K.bd = F(i++); }
      else { K.wd = K.gd = K.bd = nullptr; }
    }
    P.probe_w = F(i++); P.probe_b = F(i++);
    P.uW1 = F(i++); P.ub1 = F(i++); P.uW2 = F(i++); P.ub2 = F(i++);
    P.rW1 = F(i++); P.rb1 = F(i++); P.rW2 = F(i++); P.rb2 = F(i++);
  } else {
    // sorted-key pytree leaves: blocks, bn1_b, bn1_g, conv1_w, probe_b, probe_w, r, u
    for (int b = 0; b < 8; ++b) {
      Blk& K = P.blk[b];
      K.ds = (BLK_S[b] != 1) || (BLK_CIN[b] != BLK_COUT[b]);
      if (K.ds) {
        K.b1 = F(i++); K.b2 = F(i++); K.bd = F(i++);
        K.g1 = F(i++); K.g2 = F(i++); K.gd = F(i++);
        K.w1 = F(i++); K.w2 = F(i++); K.wd = F(i++);
      } else {
        K.b1 = F(i++); K.b2 = F(i++);
        K.g1 = F(i++); K.g2 = F(i++);
        K.w1 = F(i++); K.w2 = F(i++);
        K.wd = K.gd = K.bd = nullptr;
      }
    }
    P.bn1_b = F(i++); P.bn1_g = F(i++); P.conv1_w = F(i++);
    P.probe_b = F(i++); P.probe_w = F(i++);
    P.rW1 = F(i++); P.rW2 = F(i++); P.rb1 = F(i++); P.rb2 = F(i++);
    P.uW1 = F(i++); P.uW2 = F(i++); P.ub1 = F(i++); P.ub2 = F(i++);
    if (!xFirst) { P.x_r = F(i++); P.x_u = F(i++); }
  }
}

extern "C" void kernel_launch(void* const* d_in, const int* in_sizes, int n_in,
                              void* d_out, int out_size, void* d_ws, size_t ws_size,
                              hipStream_t stream) {
  (void)out_size; (void)ws_size;
  Net P; map_inputs(d_in, in_sizes, n_in, P);
  char* ws = (char*)d_ws;

  // ----- workspace arena (~150 MB) -----
  size_t o = 0;
  auto take = [&](size_t bytes) { size_t r = o; o += (bytes + 255) & ~size_t(255); return r; };
  auto MB = [](size_t m) { return m << 20; };
  size_t WBF  = take(MB(8));    // bf16 weight scratch
  size_t CURF = take(MB(34));   // current activation f32 (NHWC)
  size_t CURB = take(MB(17));   // bf16 mirror (NHWC, channel-padded)
  size_t T1F  = take(MB(34));   // conv scratch f32 (NHWC)
  size_t T2B  = take(MB(17));   // post-BN1 bf16 / downsample-identity f32
  size_t T3F  = take(MB(34));   // conv2 output f32 (NHWC)
  size_t STAT = take(4 * 512 * sizeof(float));
  size_t ZF   = take(256 * 512 * 4);
  size_t ZB   = take(256 * 512 * 2);
  size_t PRB  = take(100 * 256 * 4);
  size_t HARR = take(256 * 4);
  size_t NAR  = take(256 * 4);
  size_t HXF  = take(6 * 512 * 128 * 4);
  size_t HXB  = take(6 * 512 * 128 * 2);   // transposed bf16 h: [k][128][512]
  size_t LGT  = take(6 * 100 * 128 * 4);

  float* mean0 = (float*)(ws + STAT);
  float* rstd0 = mean0 + 512;
  float* mean1 = rstd0 + 512;
  float* rstd1 = mean1 + 512;

  auto cvt = [&](const float* s, unsigned short* d, int n) {
    k_f32_to_bf16<<<dim3((n + 255) / 256), dim3(256), 0, stream>>>(s, d, n);
  };
  auto conv = [&](const unsigned short* inBf, const float* w32, float* outF,
                  int Cin, int Hin, int Win, int Cout, int s, int pad, int khw) {
    int Ho = (Hin + 2 * pad - khw) / s + 1;
    int Wo = (Win + 2 * pad - khw) / s + 1;
    int Cinp = (Cin + 31) & ~31;
    int KHW2 = khw * khw;
    unsigned short* wbf = (unsigned short*)(ws + WBF);
    int wcnt = Cout * KHW2 * Cinp;
    k_w_conv_bf16<<<dim3((wcnt + 255) / 256), dim3(256), 0, stream>>>(
        w32, wbf, Cout, Cin, Cinp, KHW2);
    int G = (128 * Ho * Wo) / 64;
    dim3 g((G + 7) / 8, Cout / 64);
    k_conv_wmma<<<g, dim3(256), 0, stream>>>(inBf, wbf, outF, 128, Cinp, Hin, Win,
                                             Cout, Ho, Wo, s, pad, khw);
  };
  auto gemm = [&](const unsigned short* A, int lda, const unsigned short* B, int ldb,
                  const float* bias, int relu, float* C, int ldc, int M, int N, int K) {
    int tilesM = (M + 63) / 64;
    int G = N / 64;
    int waves = tilesM * G;
    k_gemm_wmma<<<dim3((waves + 7) / 8), dim3(256), 0, stream>>>(
        A, lda, B, ldb, bias, relu, C, ldc, M, N, K);
  };
  auto stats = [&](const float* x, int C, int HW, float* mn, float* rs) {
    k_bn_stats<<<dim3(C), dim3(256), 0, stream>>>(x, 128 * HW, C, mn, rs);
  };
  auto apply = [&](const float* x, int C, int HW, const float* g, const float* b,
                   const float* mn, const float* rs, const float* resid, int relu,
                   float* oF, unsigned short* oB) {
    int total = 128 * C * HW;
    k_bn_apply<<<dim3((total + 255) / 256), dim3(256), 0, stream>>>(
        x, total, C - 1, g, b, mn, rs, resid, relu, oF, oB);
  };

  float* CURFp = (float*)(ws + CURF);
  unsigned short* CURBp = (unsigned short*)(ws + CURB);
  float* T1Fp = (float*)(ws + T1F);
  unsigned short* T2Bp = (unsigned short*)(ws + T2B);
  float* IDNp = (float*)(ws + T2B);   // reuse after T2B consumed by conv2
  float* T3Fp = (float*)(ws + T3F);

  // ----- backbone x2 (all activations NHWC) -----
  const float* xs[2] = {P.x_u, P.x_r};
  for (int br = 0; br < 2; ++br) {
    {
      int tot = 128 * 1024 * 32;
      k_stem_nhwc<<<dim3((tot + 255) / 256), dim3(256), 0, stream>>>(xs[br], CURBp);
    }
    conv(CURBp, P.conv1_w, T1Fp, 3, 32, 32, 64, 1, 1, 3);
    stats(T1Fp, 64, 1024, mean0, rstd0);
    apply(T1Fp, 64, 1024, P.bn1_g, P.bn1_b, mean0, rstd0, nullptr, 1, CURFp, CURBp);
    int H = 32, W = 32;
    for (int bi = 0; bi < 8; ++bi) {
      const Blk& B = P.blk[bi];
      int cin = BLK_CIN[bi], cout = BLK_COUT[bi], s = BLK_S[bi];
      int Ho = H / s, Wo = W / s;
      conv(CURBp, B.w1, T1Fp, cin, H, W, cout, s, 1, 3);
      stats(T1Fp, cout, Ho * Wo, mean0, rstd0);
      apply(T1Fp, cout, Ho * Wo, B.g1, B.b1, mean0, rstd0, nullptr, 1, nullptr, T2Bp);
      conv(T2Bp, B.w2, T3Fp, cout, Ho, Wo, cout, 1, 1, 3);
      stats(T3Fp, cout, Ho * Wo, mean0, rstd0);
      const float* resid = CURFp;
      if (B.ds) {
        conv(CURBp, B.wd, T1Fp, cin, H, W, cout, s, 0, 1);
        stats(T1Fp, cout, Ho * Wo, mean1, rstd1);
        apply(T1Fp, cout, Ho * Wo, B.gd, B.bd, mean1, rstd1, nullptr, 0, IDNp, nullptr);
        resid = IDNp;
      }
      apply(T3Fp, cout, Ho * Wo, B.g2, B.b2, mean0, rstd0, resid, 1, CURFp, CURBp);
      H = Ho; W = Wo;
    }
    float* z = (float*)(ws + ZF) + br * 128 * 512;
    k_avgpool<<<dim3((128 * 512 + 255) / 256), dim3(256), 0, stream>>>(CURFp, 512, 16, z, 128);
  }

  // ----- probe + entropy + median -----
  unsigned short* zb = (unsigned short*)(ws + ZB);
  cvt((float*)(ws + ZF), zb, 256 * 512);
  unsigned short* wbf = (unsigned short*)(ws + WBF);
  cvt(P.probe_w, wbf, 100 * 512);
  gemm(wbf, 512, zb, 512, P.probe_b, 0, (float*)(ws + PRB), 256, 100, 256, 512);
  k_entropy<<<dim3(1), dim3(256), 0, stream>>>((float*)(ws + PRB), 100, 256,
                                               (float*)(ws + HARR));
  k_median_n<<<dim3(1), dim3(256), 0, stream>>>((float*)(ws + HARR), (int*)(ws + NAR));

  // ----- expert heads + gating, per branch -----
  for (int br = 0; br < 2; ++br) {
    const float* W1 = br ? P.rW1 : P.uW1;
    const float* b1 = br ? P.rb1 : P.ub1;
    const float* W2 = br ? P.rW2 : P.uW2;
    const float* b2 = br ? P.rb2 : P.ub2;
    cvt(W1, wbf, 6 * 512 * 512);
    for (int k = 0; k < 6; ++k) {
      gemm(wbf + k * 512 * 512, 512, zb + br * 128 * 512, 512,
           b1 + k * 512, 1, (float*)(ws + HXF) + k * 512 * 128, 128, 512, 128, 512);
    }
    // h [k][512][128] f32 -> hT [k][128][512] bf16 (so fc2 B is transB layout)
    {
      int tot = 6 * 512 * 128;
      k_transpose_bf16<<<dim3((tot + 255) / 256), dim3(256), 0, stream>>>(
          (float*)(ws + HXF), (unsigned short*)(ws + HXB), 6, 512, 128);
    }
    cvt(W2, wbf, 6 * 100 * 512);
    for (int k = 0; k < 6; ++k) {
      gemm(wbf + k * 100 * 512, 512, (unsigned short*)(ws + HXB) + k * 128 * 512, 512,
           b2 + k * 100, 0, (float*)(ws + LGT) + k * 100 * 128, 128, 100, 128, 512);
    }
    k_gates_mix<<<dim3(1), dim3(128), 0, stream>>>(
        (float*)(ws + LGT), (int*)(ws + NAR) + br * 128,
        (float*)d_out + br * 128 * 100, 128, 6, 100);
  }
}
